// SAE_Attention_40733469835426
// MI455X (gfx1250) — matrix operand order repaired and checked by reading, not compiled, not run on hardware
//
#include <hip/hip_runtime.h>
#include <hip/hip_bf16.h>
#include <stdint.h>

// Problem sizes
#define B_ROWS 8192
#define D_DIM  512
#define HID    8192
#define KTOP   32
#define NHEAD  4
#define HD_DIM 128

typedef __attribute__((ext_vector_type(2)))  float  v2f;
typedef __attribute__((ext_vector_type(8)))  float  v8f;
typedef __attribute__((ext_vector_type(4)))  int    v4i;
typedef __attribute__((ext_vector_type(8)))  __bf16 v8bf;
typedef __attribute__((ext_vector_type(16))) __bf16 v16bf;

// ---------------------------------------------------------------------------
// CDNA5 async global->LDS copy (ASYNCcnt-tracked), with safe fallback.
// Probe-confirmed prototype (from round-2 diagnostic):
//   void __builtin_amdgcn_global_load_async_to_lds_b128(
//       int4 __device__* gsrc, int4 __shared__* ldst, int imm_off, int cpol)
// ---------------------------------------------------------------------------
#if defined(__has_builtin)
#if __has_builtin(__builtin_amdgcn_global_load_async_to_lds_b128)
#define HAVE_ASYNC_LDS 1
#endif
#endif

#define AS1 __attribute__((address_space(1)))
#define AS3 __attribute__((address_space(3)))

#if defined(HAVE_ASYNC_LDS)
#define ASYNC_CP16(gsrc, ldst, imm) \
  __builtin_amdgcn_global_load_async_to_lds_b128( \
      (AS1 v4i*)(gsrc), (AS3 v4i*)(ldst), (imm), 0)
#endif

__device__ __forceinline__ void async_fence() {
#if defined(HAVE_ASYNC_LDS)
#if __has_builtin(__builtin_amdgcn_s_wait_asynccnt)
  __builtin_amdgcn_s_wait_asynccnt(0);
#else
  asm volatile("s_wait_asynccnt 0x0" ::: "memory");
#endif
#endif
}

__device__ __forceinline__ unsigned short f32_to_bf16_rne(float f) {
  uint32_t u = __float_as_uint(f);
  uint32_t r = u + 0x7FFFu + ((u >> 16) & 1u);
  return (unsigned short)(r >> 16);
}

// ---------------------------------------------------------------------------
// float -> bf16 (RNE) conversion
// ---------------------------------------------------------------------------
__global__ void k_f2bf(const float* __restrict__ in,
                       unsigned short* __restrict__ out, int n) {
  int i = blockIdx.x * blockDim.x + threadIdx.x;
  int stride = gridDim.x * blockDim.x;
  for (; i < n; i += stride) out[i] = f32_to_bf16_rne(in[i]);
}

// ---------------------------------------------------------------------------
// benc_eff[n] = b_enc[n] - dot(b_dec, W_enc[n,:])
// (folds the "x - b_dec" encoder input shift into the latent bias so the
//  A-tile becomes a pure byte copy for the async DMA; exact fp32 identity)
// ---------------------------------------------------------------------------
__global__ void k_benc_eff(const float* __restrict__ benc,
                           const float* __restrict__ bdec,
                           const float* __restrict__ Wenc,
                           float* __restrict__ benc_eff) {
  int n = blockIdx.x * blockDim.x + threadIdx.x;
  if (n >= HID) return;
  float s = 0.f;
  for (int k = 0; k < D_DIM; k += 4) {
    float4 w = *(const float4*)(Wenc + (size_t)n * D_DIM + k);
    float4 b = *(const float4*)(bdec + k);
    s += w.x * b.x + w.y * b.y + w.z * b.z + w.w * b.w;
  }
  benc_eff[n] = benc[n] - s;
}

// ---------------------------------------------------------------------------
// Fused encoder GEMM (exact fp32 via V_WMMA_F32_16X16X4_F32) + ReLU + top-k.
// Grid: 256 blocks = 128 M-tiles (64 rows) x 2 N-halves (4096 latents each).
// Double-buffered LDS filled by GLOBAL_LOAD_ASYNC_TO_LDS_B128; C tile staged
// in two 32-column phases to stay under 64KB static LDS.
// ---------------------------------------------------------------------------
#define EBM 64
#define EBN 64
#define EBK 32
#define ELDA 36   // padded fp32 row stride (16B aligned, conflict-free b64 reads)

__global__ __launch_bounds__(256) void k_enc_topk(
    const float* __restrict__ x, const float* __restrict__ Wenc,
    const float* __restrict__ benc_eff,
    float* __restrict__ cand_v, int* __restrict__ cand_i)
{
  __shared__ float As[2][EBM][ELDA];
  __shared__ float Bs[2][EBN][ELDA];
  __shared__ float Ct[EBM][33];           // 32 cols + pad, two phases
  __shared__ float tv[EBM][KTOP + 1];
  __shared__ int   ti[EBM][KTOP + 1];
  __shared__ float benc_s[EBN];

  const int tid  = threadIdx.x;
  const int lane = tid & 31;
  const int wid  = tid >> 5;
  const int mt = blockIdx.x >> 1;
  const int nh = blockIdx.x & 1;
  const int m0 = mt * EBM;
  const int nhalf0 = nh * (HID / 2);

  // wave tiling: 2 (M) x 4 (N) waves; wave tile = 32M x 16N
  const int wm = wid >> 2;
  const int wn = wid & 3;
  const int rb = wm * 32;
  const int l15 = lane & 15;
  const int lh  = lane >> 4;

  // cooperative fill assignment: 32B (8 floats) per thread per matrix
  const int frow = tid >> 2;
  const int fseg = (tid & 3) * 8;

  // per-row top-k running state (threads 0..63 own rows)
  int   cnt = 0, rpos = 0;
  float rmin = 0.f;

  auto enc_fill = [&](int buf, int n0, int k0) {
    const float* xs = x + (size_t)(m0 + frow) * D_DIM + k0 + fseg;
    const float* wsrc = Wenc + (size_t)(n0 + frow) * D_DIM + k0 + fseg;
#if defined(HAVE_ASYNC_LDS)
    ASYNC_CP16(xs,   &As[buf][frow][fseg], 0);
    ASYNC_CP16(xs,   &As[buf][frow][fseg], 16);
    ASYNC_CP16(wsrc, &Bs[buf][frow][fseg], 0);
    ASYNC_CP16(wsrc, &Bs[buf][frow][fseg], 16);
#else
    #pragma unroll
    for (int q = 0; q < 8; ++q) As[buf][frow][fseg + q] = xs[q];
    #pragma unroll
    for (int q = 0; q < 8; ++q) Bs[buf][frow][fseg + q] = wsrc[q];
#endif
  };

  for (int nt = 0; nt < (HID / 2) / EBN; ++nt) {
    const int n0 = nhalf0 + nt * EBN;
    v8f acc0 = {};
    v8f acc1 = {};

    int buf = 0;
    enc_fill(0, n0, 0);
    async_fence();
    __syncthreads();
    for (int kk = 0; kk < D_DIM / EBK; ++kk) {
      if (kk + 1 < D_DIM / EBK) enc_fill(buf ^ 1, n0, (kk + 1) * EBK);
      const int cb = wn * 16;
      #pragma unroll
      for (int s = 0; s < 8; ++s) {
        const int ko = s * 4 + lh * 2;
        v2f a0 = *(const v2f*)&As[buf][rb + l15][ko];
        v2f a1 = *(const v2f*)&As[buf][rb + 16 + l15][ko];
        v2f bf = *(const v2f*)&Bs[buf][cb + l15][ko];
        acc0 = __builtin_amdgcn_wmma_f32_16x16x4_f32(false, a0, false, bf,
                                                     (short)0, acc0, false, false);
        acc1 = __builtin_amdgcn_wmma_f32_16x16x4_f32(false, a1, false, bf,
                                                     (short)0, acc1, false, false);
      }
      async_fence();
      __syncthreads();
      buf ^= 1;
    }

    // ---- phase 0: columns 0..31 (waves wn==0,1 dump) ----
    if (wn < 2) {
      const int cp = wn * 16;
      #pragma unroll
      for (int v = 0; v < 8; ++v) {
        Ct[rb + v + 8 * lh][cp + l15]      = acc0[v];
        Ct[rb + 16 + v + 8 * lh][cp + l15] = acc1[v];
      }
    }
    if (tid >= 64 && tid < 128) benc_s[tid - 64] = benc_eff[n0 + (tid - 64)];
    __syncthreads();

    #pragma unroll
    for (int phase = 0; phase < 2; ++phase) {
      if (tid < EBM) {
        const int r = tid;
        for (int c = 0; c < 32; ++c) {
          float v = fmaxf(Ct[r][c] + benc_s[phase * 32 + c], 0.f);  // +b_enc_eff, ReLU
          const int col = n0 + phase * 32 + c;
          if (cnt < KTOP) {
            tv[r][cnt] = v; ti[r][cnt] = col;
            if (cnt == 0 || v < rmin) { rmin = v; rpos = cnt; }
            ++cnt;
          } else if (v > rmin) {
            tv[r][rpos] = v; ti[r][rpos] = col;
            rmin = tv[r][0]; rpos = 0;
            #pragma unroll
            for (int j = 1; j < KTOP; ++j) {
              float t = tv[r][j];
              if (t < rmin) { rmin = t; rpos = j; }
            }
          }
        }
      }
      __syncthreads();
      // ---- phase 1 dump: columns 32..63 (waves wn==2,3) ----
      if (phase == 0) {
        if (wn >= 2) {
          const int cp = (wn - 2) * 16;
          #pragma unroll
          for (int v = 0; v < 8; ++v) {
            Ct[rb + v + 8 * lh][cp + l15]      = acc0[v];
            Ct[rb + 16 + v + 8 * lh][cp + l15] = acc1[v];
          }
        }
        __syncthreads();
      }
    }
  }

  if (tid < EBM) {
    const int row = m0 + tid;
    #pragma unroll
    for (int j = 0; j < KTOP; ++j) {
      cand_v[(size_t)row * 64 + nh * KTOP + j] = tv[tid][j];
      cand_i[(size_t)row * 64 + nh * KTOP + j] = ti[tid][j];
    }
  }
}

// ---------------------------------------------------------------------------
// Merge 2x top-32 candidate lists -> final top-32 (rank select, stable ties by
// lower latent index, matching lax.top_k).
// ---------------------------------------------------------------------------
__global__ void k_topk_merge(const float* __restrict__ cand_v,
                             const int* __restrict__ cand_i,
                             float* __restrict__ vals, int* __restrict__ idx)
{
  int row = blockIdx.x * blockDim.x + threadIdx.x;
  if (row >= B_ROWS) return;
  float v[64]; int ix[64];
  #pragma unroll
  for (int j = 0; j < 64; ++j) {
    v[j]  = cand_v[(size_t)row * 64 + j];
    ix[j] = cand_i[(size_t)row * 64 + j];
  }
  #pragma unroll
  for (int a = 0; a < 64; ++a) {
    int rank = 0;
    #pragma unroll
    for (int b = 0; b < 64; ++b)
      rank += (v[b] > v[a]) || (v[b] == v[a] && ix[b] < ix[a]);
    if (rank < KTOP) {
      vals[(size_t)row * KTOP + rank] = v[a];
      idx[(size_t)row * KTOP + rank]  = ix[a];
    }
  }
}

// ---------------------------------------------------------------------------
// Sparse decode: x_recon = sum_j vals[j] * W_dec[idx[j], :] + b_dec, stored
// directly as bf16 (it only feeds bf16 GEMMs). W_dec (16MB) lives in L2.
// ---------------------------------------------------------------------------
__global__ __launch_bounds__(128) void k_decode(
    const float* __restrict__ vals, const int* __restrict__ idx,
    const float* __restrict__ Wdec, const float* __restrict__ bdec,
    unsigned short* __restrict__ xrec_bf)
{
  const int b = blockIdx.x;
  const int t = threadIdx.x;            // covers cols 4t..4t+3
  float4 acc = *(const float4*)(bdec + 4 * t);
  for (int j = 0; j < KTOP; ++j) {
    float v = vals[(size_t)b * KTOP + j];
    int   r = idx[(size_t)b * KTOP + j];
    float4 w = *(const float4*)(Wdec + (size_t)r * D_DIM + 4 * t);
    acc.x += v * w.x; acc.y += v * w.y; acc.z += v * w.z; acc.w += v * w.w;
  }
  unsigned short* o = xrec_bf + (size_t)b * D_DIM + 4 * t;
  o[0] = f32_to_bf16_rne(acc.x);
  o[1] = f32_to_bf16_rne(acc.y);
  o[2] = f32_to_bf16_rne(acc.z);
  o[3] = f32_to_bf16_rne(acc.w);
}

// ---------------------------------------------------------------------------
// bf16 GEMM: C[M,N] = A[M,K] * B[N,K]^T + bias[N], fp32 accumulate, using
// V_WMMA_F32_16X16X32_BF16. Block tile 128x128, 8 waves (4M x 2N), wave tile
// 32x64, double-buffered LDS filled by GLOBAL_LOAD_ASYNC_TO_LDS_B128.
// ---------------------------------------------------------------------------
#define GBM 128
#define GBN 128
#define GBK 32
#define GLD 40   // padded bf16 row stride (80B: 16B-aligned, conflict-free b128)

__global__ __launch_bounds__(256) void k_gemm_bf16(
    const unsigned short* __restrict__ A, const unsigned short* __restrict__ Bm,
    const float* __restrict__ bias, float* __restrict__ C,
    int M, int N, int Kd)
{
  __shared__ unsigned short As[2][GBM][GLD];
  __shared__ unsigned short Bs[2][GBN][GLD];

  const int tid  = threadIdx.x;
  const int lane = tid & 31;
  const int wid  = tid >> 5;
  const int l15  = lane & 15;
  const int lh   = lane >> 4;
  const int wm = wid >> 1;                 // 0..3 (32 rows each)
  const int wn = wid & 1;                  // 0..1 (64 cols each)
  const int bm = blockIdx.x * GBM;
  const int bn = blockIdx.y * GBN;
  const int nK = Kd / GBK;

  auto fill = [&](int buf, int k0) {
    #pragma unroll
    for (int q = 0; q < 2; ++q) {
      int task = tid + q * 256;
      int r  = task >> 2;
      int sg = (task & 3) * 8;
      const unsigned short* ga = A + (size_t)(bm + r) * Kd + k0 + sg;
      const unsigned short* gb = Bm + (size_t)(bn + r) * Kd + k0 + sg;
#if defined(HAVE_ASYNC_LDS)
      ASYNC_CP16(ga, &As[buf][r][sg], 0);
      ASYNC_CP16(gb, &Bs[buf][r][sg], 0);
#else
      *(uint4*)&As[buf][r][sg] = *(const uint4*)ga;
      *(uint4*)&Bs[buf][r][sg] = *(const uint4*)gb;
#endif
    }
#if !defined(HAVE_ASYNC_LDS)
    if (k0 + GBK < Kd) {
      __builtin_prefetch(A + (size_t)(bm + (tid >> 1)) * Kd + k0 + GBK, 0, 1);
    }
#endif
  };

  v8f acc[2][4];
  #pragma unroll
  for (int i = 0; i < 2; ++i)
    #pragma unroll
    for (int j = 0; j < 4; ++j) acc[i][j] = (v8f){};

  fill(0, 0);
  async_fence();
  __syncthreads();
  int buf = 0;
  for (int kk = 0; kk < nK; ++kk) {
    if (kk + 1 < nK) fill(buf ^ 1, (kk + 1) * GBK);

    v16bf af[2];
    #pragma unroll
    for (int i = 0; i < 2; ++i) {
      const unsigned short* p = &As[buf][wm * 32 + i * 16 + l15][8 * lh];
      v8bf lo = *(const v8bf*)p;
      v8bf hi = *(const v8bf*)(p + 16);
      af[i] = __builtin_shufflevector(lo, hi, 0,1,2,3,4,5,6,7,8,9,10,11,12,13,14,15);
    }
    v16bf bfr[4];
    #pragma unroll
    for (int j = 0; j < 4; ++j) {
      const unsigned short* p = &Bs[buf][wn * 64 + j * 16 + l15][8 * lh];
      v8bf lo = *(const v8bf*)p;
      v8bf hi = *(const v8bf*)(p + 16);
      bfr[j] = __builtin_shufflevector(lo, hi, 0,1,2,3,4,5,6,7,8,9,10,11,12,13,14,15);
    }
    #pragma unroll
    for (int i = 0; i < 2; ++i)
      #pragma unroll
      for (int j = 0; j < 4; ++j)
        acc[i][j] = __builtin_amdgcn_wmma_f32_16x16x32_bf16(
            false, af[i], false, bfr[j], (short)0, acc[i][j], false, false);

    async_fence();
    __syncthreads();
    buf ^= 1;
  }

  // epilogue: bias add + store (C element (m,n): VGPR v -> m = v + 8*lh)
  #pragma unroll
  for (int i = 0; i < 2; ++i) {
    #pragma unroll
    for (int j = 0; j < 4; ++j) {
      const int col = bn + wn * 64 + j * 16 + l15;
      const float bv = bias[col];
      const int rowb = bm + wm * 32 + i * 16 + 8 * lh;
      #pragma unroll
      for (int v = 0; v < 8; ++v)
        C[(size_t)(rowb + v) * N + col] = acc[i][j][v] + bv;
    }
  }
}

// ---------------------------------------------------------------------------
// Attention combine for position 1 only (seq len 2): one wave per (row,head).
// ---------------------------------------------------------------------------
__global__ __launch_bounds__(256) void k_attn(
    const float* __restrict__ qkv1, const float* __restrict__ kv0,
    unsigned short* __restrict__ attn_bf)
{
  const int lane = threadIdx.x & 31;
  const int wid  = threadIdx.x >> 5;
  const int lin  = blockIdx.x * 8 + wid;   // (row, head)
  const int b = lin >> 2;
  const int h = lin & 3;

  const float* q  = qkv1 + (size_t)b * 1536 + h * HD_DIM;
  const float* k1 = q + 512;
  const float* v1 = q + 1024;
  const float* k0 = kv0 + (size_t)b * 1024 + h * HD_DIM;
  const float* v0 = k0 + 512;

  float d0 = 0.f, d1 = 0.f;
  #pragma unroll
  for (int u = 0; u < 4; ++u) {
    int dd = lane + 32 * u;
    float qv = q[dd];
    d0 += qv * k0[dd];
    d1 += qv * k1[dd];
  }
  #pragma unroll
  for (int off = 16; off > 0; off >>= 1) {
    d0 += __shfl_xor(d0, off, 32);
    d1 += __shfl_xor(d1, off, 32);
  }
  const float scale = 0.08838834764831845f;  // 1/sqrt(128)
  float s0 = d0 * scale, s1 = d1 * scale;
  float mx = fmaxf(s0, s1);
  float e0 = __expf(s0 - mx), e1 = __expf(s1 - mx);
  float inv = 1.f / (e0 + e1);
  float w0 = e0 * inv, w1 = e1 * inv;

  unsigned short* o = attn_bf + (size_t)b * D_DIM + h * HD_DIM;
  #pragma unroll
  for (int u = 0; u < 4; ++u) {
    int dd = lane + 32 * u;
    o[dd] = f32_to_bf16_rne(w0 * v0[dd] + w1 * v1[dd]);
  }
}

// ---------------------------------------------------------------------------
// Host-side orchestration
// ---------------------------------------------------------------------------
extern "C" void kernel_launch(void* const* d_in, const int* in_sizes, int n_in,
                              void* d_out, int out_size, void* d_ws, size_t ws_size,
                              hipStream_t stream)
{
  (void)in_sizes; (void)n_in; (void)out_size; (void)ws_size;
  const float* x    = (const float*)d_in[0];
  const float* Wenc = (const float*)d_in[1];
  const float* benc = (const float*)d_in[2];
  const float* Wdec = (const float*)d_in[3];
  const float* bdec = (const float*)d_in[4];
  const float* Win  = (const float*)d_in[5];
  const float* bin  = (const float*)d_in[6];
  const float* Wout = (const float*)d_in[7];
  const float* bout = (const float*)d_in[8];
  float* out = (float*)d_out;

  char* ws = (char*)d_ws;
  size_t off = 0;
  auto carve = [&](size_t bytes) { char* p = ws + off; off += (bytes + 255) & ~(size_t)255; return p; };

  unsigned short* x_bf     = (unsigned short*)carve((size_t)B_ROWS * D_DIM * 2);    // 8 MB
  unsigned short* win_bf   = (unsigned short*)carve((size_t)3 * D_DIM * D_DIM * 2); // 1.5 MB
  unsigned short* wout_bf  = (unsigned short*)carve((size_t)D_DIM * D_DIM * 2);     // 0.5 MB
  float*          benc_eff = (float*)carve((size_t)HID * 4);                        // 32 KB
  float*          cand_v   = (float*)carve((size_t)B_ROWS * 64 * 4);                // 2 MB
  int*            cand_i   = (int*)carve((size_t)B_ROWS * 64 * 4);                  // 2 MB
  float*          vals     = (float*)carve((size_t)B_ROWS * KTOP * 4);              // 1 MB
  int*            idx      = (int*)carve((size_t)B_ROWS * KTOP * 4);                // 1 MB
  unsigned short* xrec_bf  = (unsigned short*)carve((size_t)B_ROWS * D_DIM * 2);    // 8 MB
  unsigned short* attn_bf  = (unsigned short*)carve((size_t)B_ROWS * D_DIM * 2);    // 8 MB
  float*          qkv1     = (float*)carve((size_t)B_ROWS * 1536 * 4);              // 48 MB
  float*          kv0      = (float*)carve((size_t)B_ROWS * 1024 * 4);              // 32 MB

  // bf16 conversions + effective latent bias
  k_f2bf<<<2048, 256, 0, stream>>>(x, x_bf, B_ROWS * D_DIM);
  k_f2bf<<<768, 256, 0, stream>>>(Win, win_bf, 3 * D_DIM * D_DIM);
  k_f2bf<<<256, 256, 0, stream>>>(Wout, wout_bf, D_DIM * D_DIM);
  k_benc_eff<<<HID / 256, 256, 0, stream>>>(benc, bdec, Wenc, benc_eff);

  // fused encoder GEMM (fp32 WMMA) + ReLU + per-half top-32
  k_enc_topk<<<256, 256, 0, stream>>>(x, Wenc, benc_eff, cand_v, cand_i);

  // merge candidate halves -> final top-32
  k_topk_merge<<<B_ROWS / 256, 256, 0, stream>>>(cand_v, cand_i, vals, idx);

  // sparse decode -> x_recon (bf16)
  k_decode<<<B_ROWS, 128, 0, stream>>>(vals, idx, Wdec, bdec, xrec_bf);

  // qkv for position 1: x_recon @ W_in^T -> [B, 1536] (q1|k1|v1)
  k_gemm_bf16<<<dim3(B_ROWS / GBM, 1536 / GBN), 256, 0, stream>>>(
      xrec_bf, win_bf, bin, qkv1, B_ROWS, 1536, D_DIM);

  // k,v for position 0: x @ W_in[512:1536]^T -> [B, 1024] (k0|v0)
  k_gemm_bf16<<<dim3(B_ROWS / GBM, 1024 / GBN), 256, 0, stream>>>(
      x_bf, win_bf + (size_t)D_DIM * D_DIM, bin + D_DIM, kv0, B_ROWS, 1024, D_DIM);

  // softmax combine (position 1) -> attn_bf
  k_attn<<<B_ROWS * NHEAD / 8, 256, 0, stream>>>(qkv1, kv0, attn_bf);

  // out projection: attn @ W_out^T + b_out -> d_out (fp32)
  k_gemm_bf16<<<dim3(B_ROWS / GBM, D_DIM / GBN), 256, 0, stream>>>(
      attn_bf, wout_bf, bout, out, B_ROWS, D_DIM, D_DIM);
}
